// seq2seq_26499948216460
// MI455X (gfx1250) — matrix operand order repaired and checked by reading, not compiled, not run on hardware
//
#include <hip/hip_runtime.h>
#include <cmath>

// ---------------------------------------------------------------------------
// Seq2seq GRU encoder/decoder + Bahdanau attention for gfx1250 (MI455X).
// Heavy GEMMs: v_wmma_f32_16x16x32_bf16, B panel staged to LDS per block via
// the Tensor Data Mover (tensor_load_to_lds + s_wait_tensorcnt).
// ---------------------------------------------------------------------------

typedef __attribute__((ext_vector_type(16))) __bf16       v16bf;
typedef __attribute__((ext_vector_type(8)))  float        v8f;
typedef __attribute__((ext_vector_type(4)))  unsigned int u32x4;
typedef __attribute__((ext_vector_type(8)))  int          i32x8;
typedef __attribute__((ext_vector_type(4)))  int          i32x4;

#if defined(__gfx1250__) && __has_builtin(__builtin_amdgcn_tensor_load_to_lds)
#define USE_TDM 1
#else
#define USE_TDM 0
#endif

__device__ __forceinline__ float sigmoidf_(float x) {
    return 1.0f / (1.0f + __expf(-x));
}

// ---------------------------------------------------------------------------
// Generic GEMM: C[M,N] = A[M,K] @ B[K,N] (+ bias[N]).
// Optional embedding gather on A rows (ids != nullptr).
// Grid: x = N/16 tile columns, y = M strips of (blockDim/32) tiles.
// Each block stages a 32x16 fp32 B tile in LDS (TDM DMA issued by wave 0),
// shared by all waves; each active wave computes one 16x16 output tile.
// K % 32 == 0, N % 16 == 0 required (true for all call sites here).
// ---------------------------------------------------------------------------
__global__ void wmma_gemm_bf16(const float* __restrict__ A, int lda,
                               const int* __restrict__ ids, int ids_ld, int rows_per_b,
                               const float* __restrict__ B, int ldb,
                               const float* __restrict__ bias,
                               float* __restrict__ C, int ldc,
                               int M, int N, int K)
{
    __shared__ float Bs[32 * 16];            // one K-chunk of the B panel

    const int lane   = (int)(threadIdx.x & 31);
    const int waveId = (int)(threadIdx.x >> 5);
    const int tile_n = (int)blockIdx.x;
    const int tile_m = (int)blockIdx.y * (int)(blockDim.x >> 5) + waveId;
    const bool active = (tile_m * 16 < M);   // wave-uniform

    const float* Arow = A;
    if (active) {
        int mrow = tile_m * 16 + (lane & 15);
        if (ids) {
            int bb = mrow / rows_per_b;
            int tt = mrow - bb * rows_per_b;
            Arow = A + (size_t)ids[bb * ids_ld + tt] * (size_t)lda;
        } else {
            Arow = A + (size_t)mrow * (size_t)lda;
        }
    }

    // A fragment (16-bit 16x32): lanes 0-15 -> K {0..7,16..23}; 16-31 -> K {8..15,24..31}
    const int kbA  = (lane < 16) ? 0 : 8;
    // B fragment (32x16): lanes 0-15 -> K {0..7,16..23}; 16-31 -> K {8..15,24..31}
    const int kbB  = (lane < 16) ? 0 : 8;
    const int ncol = tile_n * 16 + (lane & 15);

    v8f acc = {0.f, 0.f, 0.f, 0.f, 0.f, 0.f, 0.f, 0.f};

#if USE_TDM
    const unsigned lds_off = (unsigned)(size_t)(void*)&Bs[0];  // low 32 bits = LDS offset
#endif

    for (int k0 = 0; k0 < K; k0 += 32) {
        __syncthreads();                     // previous chunk fully consumed

#if USE_TDM
        if (waveId == 0) {
            // 2D tile: 16 elements wide (contiguous), 32 rows, row stride = ldb.
            unsigned long long gaddr =
                (unsigned long long)(size_t)(B + (size_t)k0 * (size_t)ldb + tile_n * 16);
            u32x4 g0;
            g0[0] = 1u;                                            // count=1, user mode
            g0[1] = lds_off;                                       // lds_addr
            g0[2] = (unsigned)(gaddr & 0xFFFFFFFFu);               // global_addr[31:0]
            g0[3] = (unsigned)((gaddr >> 32) & 0x01FFFFFFu)        // global_addr[56:32]
                  | (2u << 30);                                    // type=2 (image)
            i32x8 g1;
            g1[0] = (2 << 16);                                     // data_size=4B; mask=0
            g1[1] = (16 << 16);                                    // tensor_dim0 = 16
            g1[2] = (32 << 16);                                    // tensor_dim1 = 32
            g1[3] = (16 << 16);                                    // tile_dim0 = 16
            g1[4] = 32;                                            // tile_dim1 = 32
            g1[5] = ldb;                                           // tensor_dim0_stride
            g1[6] = 0;
            g1[7] = 0;
            i32x4 gz = {0, 0, 0, 0};
#if __clang_major__ >= 23
            i32x8 gz8 = {0, 0, 0, 0, 0, 0, 0, 0};
            __builtin_amdgcn_tensor_load_to_lds(g0, g1, gz, gz, gz8, 0);
#else
            __builtin_amdgcn_tensor_load_to_lds(g0, g1, gz, gz, 0);
#endif
            __builtin_amdgcn_s_wait_tensorcnt(0);
        }
#else
        {
            const float* Bp = B + (size_t)k0 * (size_t)ldb + tile_n * 16;
            for (int i = (int)threadIdx.x; i < 32 * 16; i += (int)blockDim.x)
                Bs[i] = Bp[(size_t)(i >> 4) * (size_t)ldb + (i & 15)];
        }
#endif
        __syncthreads();                     // B tile visible to all waves

        if (active) {
            const float4* ap4 = (const float4*)(Arow + k0 + kbA);
            float4 a0 = ap4[0];              // K = kbA + 0..3
            float4 a1 = ap4[1];              // K = kbA + 4..7
            float4 a2 = ap4[4];              // K = kbA + 16..19
            float4 a3 = ap4[5];              // K = kbA + 20..23

            v16bf afrag;
            afrag[0]  = (__bf16)a0.x;  afrag[1]  = (__bf16)a0.y;
            afrag[2]  = (__bf16)a0.z;  afrag[3]  = (__bf16)a0.w;
            afrag[4]  = (__bf16)a1.x;  afrag[5]  = (__bf16)a1.y;
            afrag[6]  = (__bf16)a1.z;  afrag[7]  = (__bf16)a1.w;
            afrag[8]  = (__bf16)a2.x;  afrag[9]  = (__bf16)a2.y;
            afrag[10] = (__bf16)a2.z;  afrag[11] = (__bf16)a2.w;
            afrag[12] = (__bf16)a3.x;  afrag[13] = (__bf16)a3.y;
            afrag[14] = (__bf16)a3.z;  afrag[15] = (__bf16)a3.w;

            v16bf bfrag;
#pragma unroll
            for (int i = 0; i < 16; ++i)
                bfrag[i] = (__bf16)Bs[(kbB + (i & 7) + ((i >> 3) << 4)) * 16 + (lane & 15)];

            acc = __builtin_amdgcn_wmma_f32_16x16x32_bf16(
                      false, afrag, false, bfrag, (short)0, acc, false, false);
        }
    }

    if (active) {
        const float badd = bias ? bias[ncol] : 0.0f;
        const int row0 = tile_m * 16 + ((lane < 16) ? 0 : 8);
#pragma unroll
        for (int v = 0; v < 8; ++v)
            C[(size_t)(row0 + v) * (size_t)ldc + ncol] = acc[v] + badd;
    }
}

// ---------------------------------------------------------------------------
// GRU gate update (gate order z, r, h; Keras convention):
//   z  = sigmoid(xt[j]      + hz[j])
//   r  = sigmoid(xt[H+j]    + hz[H+j])
//   hh = tanh   (xt[2H+j] + r*hz[2H+j])
//   h' = z*h + (1-z)*hh
// ---------------------------------------------------------------------------
__global__ void gru_gate(const float* __restrict__ xz,   // [B, Tsteps, 3H]
                         const float* __restrict__ hz,   // [B, 3H]
                         float* __restrict__ h,          // [B, H]  (in/out)
                         float* __restrict__ out,        // [B, Tsteps, H]
                         int t, int Tsteps)
{
    const int H = 1024;
    int idx = (int)(blockIdx.x * blockDim.x + threadIdx.x);
    if (idx >= 32 * H) return;
    int j = idx & (H - 1);
    int b = idx >> 10;
    const float* xt  = xz + ((size_t)b * Tsteps + t) * (size_t)(3 * H);
    const float* hzb = hz + (size_t)b * (3 * H);
    float z  = sigmoidf_(xt[j]         + hzb[j]);
    float r  = sigmoidf_(xt[H + j]     + hzb[H + j]);
    float hh = tanhf    (xt[2 * H + j] + r * hzb[2 * H + j]);
    float hn = z * h[idx] + (1.0f - z) * hh;
    h[idx] = hn;
    out[((size_t)b * Tsteps + t) * (size_t)H + j] = hn;
}

// ---------------------------------------------------------------------------
// Attention scores: s[b,d,e] = sum_k tanh(w1e[b,e,k] + w2d[b,d,k]) * Vw[k] + Vb
// One wave per (b,d,e); 32-lane strided reduction + shuffle tree.
// ---------------------------------------------------------------------------
__global__ void attn_scores(const float* __restrict__ w1e,   // [B, 64, H]
                            const float* __restrict__ w2d,   // [B, 63, H]
                            const float* __restrict__ Vw,    // [H]
                            const float* __restrict__ Vb,    // [1]
                            float* __restrict__ scores)      // [B, 63, 64]
{
    const int NB = 32, ND = 63, NE = 64, H = 1024;
    int wave = (int)((blockIdx.x * blockDim.x + threadIdx.x) >> 5);
    int lane = (int)(threadIdx.x & 31);
    if (wave >= NB * ND * NE) return;
    int e  = wave & (NE - 1);
    int bd = wave >> 6;
    int d  = bd % ND;
    int b  = bd / ND;
    const float* p1 = w1e + ((size_t)b * NE + e) * (size_t)H;
    const float* p2 = w2d + ((size_t)b * ND + d) * (size_t)H;
    float acc = 0.0f;
    for (int k = lane; k < H; k += 32)
        acc += tanhf(p1[k] + p2[k]) * Vw[k];
#pragma unroll
    for (int off = 16; off > 0; off >>= 1)
        acc += __shfl_xor(acc, off, 32);
    if (lane == 0) scores[wave] = acc + Vb[0];
}

// ---------------------------------------------------------------------------
// Softmax over the encoder axis (64 values) per (b,d) row. In-place.
// ---------------------------------------------------------------------------
__global__ void softmax64(float* __restrict__ s)   // [B*63, 64]
{
    int r = (int)(blockIdx.x * blockDim.x + threadIdx.x);
    if (r >= 32 * 63) return;
    float* p = s + (size_t)r * 64;
    float m = -3.402823466e+38f;
    for (int e = 0; e < 64; ++e) m = fmaxf(m, p[e]);
    float sum = 0.0f;
    for (int e = 0; e < 64; ++e) { float ex = __expf(p[e] - m); p[e] = ex; sum += ex; }
    float inv = 1.0f / sum;
    for (int e = 0; e < 64; ++e) p[e] *= inv;
}

// ---------------------------------------------------------------------------
// context[b,d,h] = sum_e attn[b,d,e] * enc_out[b,e,h]
// av[b,d, 0:H]  = tanh(context);  av[b,d, H:2H] = tanh(dec_out)
// ---------------------------------------------------------------------------
__global__ void context_av(const float* __restrict__ attn,     // [B,63,64]
                           const float* __restrict__ enc_out,  // [B,64,H]
                           const float* __restrict__ dec_out,  // [B,63,H]
                           float* __restrict__ av)             // [B,63,2H]
{
    const int ND = 63, NE = 64, H = 1024;
    int idx = (int)(blockIdx.x * blockDim.x + threadIdx.x);
    if (idx >= 32 * ND * H) return;
    int hh = idx & (H - 1);
    int bd = idx >> 10;
    int d  = bd % ND;
    int b  = bd / ND;
    const float* ap = attn + ((size_t)b * ND + d) * NE;
    const float* ep = enc_out + (size_t)b * NE * H + hh;
    float ctx = 0.0f;
#pragma unroll 4
    for (int e = 0; e < NE; ++e)
        ctx += ap[e] * ep[(size_t)e * H];
    float* o = av + ((size_t)b * ND + d) * (size_t)(2 * H);
    o[hh]     = tanhf(ctx);
    o[H + hh] = tanhf(dec_out[((size_t)b * ND + d) * (size_t)H + hh]);
}

// ---------------------------------------------------------------------------
// Host-side launch
// ---------------------------------------------------------------------------
static inline void launch_gemm(const float* A, int lda,
                               const int* ids, int ids_ld, int rpb,
                               const float* B, int ldb, const float* bias,
                               float* C, int ldc, int M, int N, int K,
                               hipStream_t s)
{
    int tilesM = (M + 15) / 16;
    int tilesN = N / 16;
    int wpb    = tilesM < 8 ? tilesM : 8;     // waves per block
    dim3 grid(tilesN, (tilesM + wpb - 1) / wpb);
    dim3 block(32 * wpb);
    wmma_gemm_bf16<<<grid, block, 0, s>>>(A, lda, ids, ids_ld, rpb,
                                          B, ldb, bias, C, ldc, M, N, K);
}

extern "C" void kernel_launch(void* const* d_in, const int* in_sizes, int n_in,
                              void* d_out, int out_size, void* d_ws, size_t ws_size,
                              hipStream_t stream)
{
    (void)in_sizes; (void)n_in; (void)out_size; (void)ws_size;

    const int B = 32, Tenc = 64, Tdec = 63, E = 256, H = 1024, H3 = 3072, V = 16000;

    const int*   input_ids  = (const int*)  d_in[0];
    const int*   target_ids = (const int*)  d_in[1];
    const float* emb_enc    = (const float*)d_in[2];
    const float* emb_dec    = (const float*)d_in[3];
    const float* enc_Wx     = (const float*)d_in[4];
    const float* enc_Wh     = (const float*)d_in[5];
    const float* enc_b      = (const float*)d_in[6];
    const float* dec_Wx     = (const float*)d_in[7];
    const float* dec_Wh     = (const float*)d_in[8];
    const float* dec_b      = (const float*)d_in[9];
    const float* W1         = (const float*)d_in[10];
    const float* W2         = (const float*)d_in[11];
    const float* V_w        = (const float*)d_in[12];
    const float* V_b        = (const float*)d_in[13];
    const float* Wp         = (const float*)d_in[14];
    const float* bp         = (const float*)d_in[15];
    float*       logits     = (float*)d_out;

    // ---- workspace carve (floats) ----
    float* ws      = (float*)d_ws;
    float* xz_enc  = ws;  ws += (size_t)B * Tenc * H3;   // 6,291,456
    float* xz_dec  = ws;  ws += (size_t)B * Tdec * H3;   // 6,193,152
    float* hz      = ws;  ws += (size_t)B * H3;          //    98,304
    float* h       = ws;  ws += (size_t)B * H;           //    32,768
    float* enc_out = ws;  ws += (size_t)B * Tenc * H;    // 2,097,152
    float* dec_out = ws;  ws += (size_t)B * Tdec * H;    // 2,064,384
    float* w1e     = ws;  ws += (size_t)B * Tenc * H;    // 2,097,152
    float* w2d     = ws;  ws += (size_t)B * Tdec * H;    // 2,064,384
    float* attn    = ws;  ws += (size_t)B * Tdec * Tenc; //   129,024
    float* av      = ws;  ws += (size_t)B * Tdec * 2*H;  // 4,128,768
    // total ~ 25.2M floats (~101 MB)

    // h0 = 0
    hipMemsetAsync(h, 0, (size_t)B * H * sizeof(float), stream);

    // xz_enc = emb_enc[input_ids] @ enc_Wx + enc_b   (gathered-A GEMM)
    launch_gemm(emb_enc, E, input_ids, Tenc, Tenc,
                enc_Wx, H3, enc_b, xz_enc, H3, B * Tenc, H3, E, stream);
    // xz_dec = emb_dec[target_ids[:, :-1]] @ dec_Wx + dec_b
    launch_gemm(emb_dec, E, target_ids, Tenc, Tdec,
                dec_Wx, H3, dec_b, xz_dec, H3, B * Tdec, H3, E, stream);

    // ---- encoder GRU ----
    for (int t = 0; t < Tenc; ++t) {
        launch_gemm(h, H, nullptr, 0, 0, enc_Wh, H3, nullptr,
                    hz, H3, B, H3, H, stream);
        gru_gate<<<(B * H + 255) / 256, 256, 0, stream>>>(xz_enc, hz, h, enc_out, t, Tenc);
    }

    // ---- decoder GRU (h currently holds enc_state) ----
    for (int t = 0; t < Tdec; ++t) {
        launch_gemm(h, H, nullptr, 0, 0, dec_Wh, H3, nullptr,
                    hz, H3, B, H3, H, stream);
        gru_gate<<<(B * H + 255) / 256, 256, 0, stream>>>(xz_dec, hz, h, dec_out, t, Tdec);
    }

    // ---- attention ----
    launch_gemm(enc_out, H, nullptr, 0, 0, W1, H, nullptr,
                w1e, H, B * Tenc, H, H, stream);
    launch_gemm(dec_out, H, nullptr, 0, 0, W2, H, nullptr,
                w2d, H, B * Tdec, H, H, stream);

    {
        int waves   = B * Tdec * Tenc;           // 129,024 waves
        int threads = 256;
        int blocks  = (waves + 7) / 8;
        attn_scores<<<blocks, threads, 0, stream>>>(w1e, w2d, V_w, V_b, attn);
    }
    softmax64<<<(B * Tdec + 255) / 256, 256, 0, stream>>>(attn);

    context_av<<<(B * Tdec * H + 255) / 256, 256, 0, stream>>>(attn, enc_out, dec_out, av);

    // ---- logits = av @ Wp + bp  (the big, HBM-bound GEMM) ----
    launch_gemm(av, 2 * H, nullptr, 0, 0, Wp, V, bp,
                logits, V, B * Tdec, V, 2 * H, stream);
}
